// BaseMoELayer_81853486727438
// MI455X (gfx1250) — matrix operand order repaired and checked by reading, not compiled, not run on hardware
//
#include <hip/hip_runtime.h>
#include <hip/hip_bf16.h>

// ---------------- problem constants ----------------
constexpr int D = 2048;    // model dim
constexpr int E = 8;       // experts
constexpr int F = 1024;    // expert hidden
constexpr int MT = 64;     // tokens per expert-block (4 WMMA row tiles)
constexpr int XS = D + 8;  // LDS row stride (halves) -> 4-bank shift per row

typedef __attribute__((ext_vector_type(16))) _Float16 v16h;
typedef __attribute__((ext_vector_type(8)))  _Float16 v8h;
typedef __attribute__((ext_vector_type(8)))  float    v8f;
typedef __attribute__((ext_vector_type(4)))  int      v4i;

#define AS_GLOBAL __attribute__((address_space(1)))
#define AS_LDS    __attribute__((address_space(3)))

__device__ __forceinline__ v16h cat8(v8h lo, v8h hi) {
  return __builtin_shufflevector(lo, hi, 0,1,2,3,4,5,6,7,8,9,10,11,12,13,14,15);
}
// A-fragment (16x32 f16): lane l holds row l%16, 16 consecutive K at +(l<16?0:16).
// Caller pre-adds that offset; chunks are ptr[0..7] and ptr[8..15].
__device__ __forceinline__ v16h loadA(const _Float16* p) {
  v8h lo = *(const v8h*)p;
  v8h hi = *(const v8h*)(p + 8);
  return cat8(lo, hi);
}
// B-fragment (32x16 f16): lane l holds col l%16, 16 consecutive K at +(l<16?0:16).
// Caller pre-adds that offset; here just 16 contiguous halves.
__device__ __forceinline__ v16h loadB(const _Float16* p) {
  v8h lo = *(const v8h*)p;
  v8h hi = *(const v8h*)(p + 8);
  return cat8(lo, hi);
}
__device__ __forceinline__ v8f wmma16(v16h a, v16h b, v8f c) {
  return __builtin_amdgcn_wmma_f32_16x16x32_f16(false, a, false, b, (short)0, c, false, false);
}

// ---------------- utility kernels ----------------
__global__ void fill0_kernel(unsigned int* __restrict__ p, size_t n) {
  size_t i = (size_t)blockIdx.x * blockDim.x + threadIdx.x;
  size_t stride = (size_t)gridDim.x * blockDim.x;
  for (; i < n; i += stride) p[i] = 0u;
}

__global__ void cvt_f16_kernel(const float* __restrict__ in, _Float16* __restrict__ out, size_t n) {
  size_t i = (size_t)blockIdx.x * blockDim.x + threadIdx.x;
  size_t stride = (size_t)gridDim.x * blockDim.x;
  for (; i < n; i += stride) out[i] = (_Float16)in[i];
}

// [batch][A][B] f32 -> [batch][B][A] f16 (tiled, A and B multiples of 32)
__global__ void transpose_cvt_kernel(const float* __restrict__ in, _Float16* __restrict__ out,
                                     int A, int B) {
  __shared__ _Float16 tile[32][33];
  size_t boff = (size_t)blockIdx.z * A * B;
  int a0 = blockIdx.y * 32, b0 = blockIdx.x * 32;
  for (int r = threadIdx.y; r < 32; r += 8)
    tile[r][threadIdx.x] = (_Float16)in[boff + (size_t)(a0 + r) * B + b0 + threadIdx.x];
  __syncthreads();
  for (int r = threadIdx.y; r < 32; r += 8)
    out[boff + (size_t)(b0 + r) * A + a0 + threadIdx.x] = tile[threadIdx.x][r];
}

// ---------------- gating: one wave32 per token ----------------
__global__ __launch_bounds__(128) void gate_kernel(
    const float* __restrict__ x, const float* __restrict__ wg,
    int* __restrict__ cnt, float* __restrict__ imp,
    int* __restrict__ atok, float* __restrict__ asc, int N)
{
  int wid = threadIdx.x >> 5;
  int lane = threadIdx.x & 31;
  int t = blockIdx.x * 4 + wid;
  if (t >= N) return;

  float acc[E];
#pragma unroll
  for (int e = 0; e < E; ++e) acc[e] = 0.f;
  const float* xp = x + (size_t)t * D;
  for (int d = lane; d < D; d += 32) {
    float xv = xp[d];
#pragma unroll
    for (int e = 0; e < E; ++e) acc[e] += xv * wg[d * E + e];
  }
#pragma unroll
  for (int e = 0; e < E; ++e)
    for (int off = 16; off > 0; off >>= 1) acc[e] += __shfl_xor(acc[e], off, 32);

  if (lane == 0) {
    float mx = acc[0];
#pragma unroll
    for (int e = 1; e < E; ++e) mx = fmaxf(mx, acc[e]);
    float p[E], s = 0.f;
#pragma unroll
    for (int e = 0; e < E; ++e) { p[e] = __expf(acc[e] - mx); s += p[e]; }
    float invs = 1.f / s;
#pragma unroll
    for (int e = 0; e < E; ++e) p[e] *= invs;
    int i0 = 0;
#pragma unroll
    for (int e = 1; e < E; ++e) if (p[e] > p[i0]) i0 = e;
    int i1 = (i0 == 0) ? 1 : 0;
#pragma unroll
    for (int e = 0; e < E; ++e) if (e != i0 && p[e] > p[i1]) i1 = e;
    float v0 = p[i0], v1 = p[i1];
    float inv = 1.f / (v0 + v1 + 1e-10f);
    v0 *= inv; v1 *= inv;
    int p0 = atomicAdd(&cnt[i0], 1);
    atok[(size_t)i0 * N + p0] = t;  asc[(size_t)i0 * N + p0] = v0;
    int p1 = atomicAdd(&cnt[i1], 1);
    atok[(size_t)i1 * N + p1] = t;  asc[(size_t)i1 * N + p1] = v1;
    atomicAdd(&imp[i0], v0);
    atomicAdd(&imp[i1], v1);
  }
}

// per-expert h-buffer offsets: prefix over counts rounded up to MT (pad-safe)
__global__ void prefix_kernel(const int* __restrict__ cnt, int* __restrict__ hoff) {
  if (threadIdx.x == 0 && blockIdx.x == 0) {
    int s = 0;
#pragma unroll
    for (int e = 0; e < E; ++e) {
      hoff[e] = s;
      s += ((cnt[e] + MT - 1) / MT) * MT;
    }
  }
}

// ---------------- routed SwiGLU experts, WMMA f16->f32 ----------------
// block = (64-token tile of expert blockIdx.y), 8 waves; x tile in LDS,
// h streamed through private global f16 scratch (L0/L2 resident).
__global__ __launch_bounds__(256) void moe_kernel(
    const _Float16* __restrict__ xh,  const _Float16* __restrict__ w1t,
    const _Float16* __restrict__ w3t, const _Float16* __restrict__ w2t,
    const int* __restrict__ atok, const float* __restrict__ asc,
    const int* __restrict__ cnt, const int* __restrict__ hoff,
    _Float16* __restrict__ hb, float* __restrict__ y, int N)
{
  int e = blockIdx.y;
  int count = cnt[e];
  int base = blockIdx.x * MT;
  if (base >= count) return;

  extern __shared__ _Float16 xs[];   // [MT][XS]
  __shared__ int   tok[MT];
  __shared__ float sc[MT];

  int tid = threadIdx.x;
  if (tid < MT) {
    int i = base + tid;
    tok[tid] = (i < count) ? atok[(size_t)e * N + i] : -1;
    sc[tid]  = (i < count) ? asc[(size_t)e * N + i] : 0.f;
  }
  __syncthreads();

  // ---- stage x tile into LDS: 4 threads per row, async-to-LDS if available ----
  {
    int r = tid >> 2;
    int t = tok[r];
    if (t < 0) t = 0;  // pad rows: load finite data, gated at output by tok<0
    const _Float16* src = xh + (size_t)t * D + (tid & 3) * 8;
    _Float16*       dst = xs + r * XS + (tid & 3) * 8;
#if __has_builtin(__builtin_amdgcn_global_load_async_to_lds_b128)
    for (int c = 0; c < D; c += 32)
      __builtin_amdgcn_global_load_async_to_lds_b128(
          (AS_GLOBAL v4i*)(src + c),
          (AS_LDS    v4i*)(dst + c), 0, 0);
#if __has_builtin(__builtin_amdgcn_s_wait_asynccnt)
    __builtin_amdgcn_s_wait_asynccnt(0);
#else
    asm volatile("s_wait_asynccnt 0x0" ::: "memory");
#endif
#else
    for (int c = 0; c < D; c += 32)
      *(v8h*)(dst + c) = *(const v8h*)(src + c);
#endif
  }
  __syncthreads();

  int wave = tid >> 5, lane = tid & 31;
  int ln = lane & 15;
  int hiHalf = lane >> 4;

  // ---- stage 1: g = x@W1, u = x@W3, h = silu(g)*u -> hb ----
  const _Float16* w1e = w1t + (size_t)e * F * D;
  const _Float16* w3e = w3t + (size_t)e * F * D;
  _Float16* hrow = hb + ((size_t)hoff[e] + base) * F;   // this block's [MT][F] region
  const _Float16* aB[4];
#pragma unroll
  for (int rt = 0; rt < 4; ++rt) aB[rt] = xs + (rt * 16 + ln) * XS + hiHalf * 16;

  for (int ft = wave; ft < F / 16; ft += 8) {
    v8f g[4] = {};
    v8f u[4] = {};
    const _Float16* b1 = w1e + (size_t)(ft * 16 + ln) * D + hiHalf * 16;
    const _Float16* b3 = w3e + (size_t)(ft * 16 + ln) * D + hiHalf * 16;
    __builtin_prefetch(w1e + (size_t)((ft + 8) * 16 + ln) * D, 0, 1);
    __builtin_prefetch(w3e + (size_t)((ft + 8) * 16 + ln) * D, 0, 1);
    for (int k = 0; k < D; k += 32) {
      v16h a0 = loadA(aB[0] + k);
      v16h a1 = loadA(aB[1] + k);
      v16h a2 = loadA(aB[2] + k);
      v16h a3 = loadA(aB[3] + k);
      v16h bf = loadB(b1 + k);
      g[0] = wmma16(a0, bf, g[0]);
      g[1] = wmma16(a1, bf, g[1]);
      g[2] = wmma16(a2, bf, g[2]);
      g[3] = wmma16(a3, bf, g[3]);
      bf = loadB(b3 + k);
      u[0] = wmma16(a0, bf, u[0]);
      u[1] = wmma16(a1, bf, u[1]);
      u[2] = wmma16(a2, bf, u[2]);
      u[3] = wmma16(a3, bf, u[3]);
    }
    int col = ft * 16 + ln;
#pragma unroll
    for (int rt = 0; rt < 4; ++rt) {
#pragma unroll
      for (int r = 0; r < 8; ++r) {
        int row = rt * 16 + r + hiHalf * 8;
        float gv = g[rt][r];
        float hv = gv / (1.f + __expf(-gv)) * u[rt][r];
        hrow[(size_t)row * F + col] = (_Float16)hv;
      }
    }
  }
  __threadfence_block();
  __syncthreads();

  // ---- stage 2: y += score * (h @ W2) ----
  const _Float16* w2e = w2t + (size_t)e * D * F;
  const _Float16* hB[4];
#pragma unroll
  for (int rt = 0; rt < 4; ++rt) hB[rt] = hrow + (size_t)(rt * 16 + ln) * F + hiHalf * 16;

  for (int dt = wave; dt < D / 16; dt += 8) {
    v8f yv[4] = {};
    const _Float16* b2 = w2e + (size_t)(dt * 16 + ln) * F + hiHalf * 16;
    __builtin_prefetch(w2e + (size_t)((dt + 8) * 16 + ln) * F, 0, 1);
    for (int k = 0; k < F; k += 32) {
      v16h bf = loadB(b2 + k);
      yv[0] = wmma16(loadA(hB[0] + k), bf, yv[0]);
      yv[1] = wmma16(loadA(hB[1] + k), bf, yv[1]);
      yv[2] = wmma16(loadA(hB[2] + k), bf, yv[2]);
      yv[3] = wmma16(loadA(hB[3] + k), bf, yv[3]);
    }
    int col = dt * 16 + ln;
#pragma unroll
    for (int rt = 0; rt < 4; ++rt) {
#pragma unroll
      for (int r = 0; r < 8; ++r) {
        int row = rt * 16 + r + hiHalf * 8;
        int t = tok[row];
        if (t >= 0) unsafeAtomicAdd(&y[(size_t)t * D + col], yv[rt][r] * sc[row]);
      }
    }
  }
}

// ---------------- balance loss + aux outputs ----------------
__global__ void finalize_kernel(const int* __restrict__ cnt, const float* __restrict__ imp,
                                float* __restrict__ tail) {
  if (threadIdx.x != 0 || blockIdx.x != 0) return;
  float iv[E], lv[E];
  float mi = 0.f, ml = 0.f;
#pragma unroll
  for (int e = 0; e < E; ++e) { iv[e] = imp[e]; lv[e] = (float)cnt[e]; mi += iv[e]; ml += lv[e]; }
  mi /= E; ml /= E;
  float viv = 0.f, vlv = 0.f;
#pragma unroll
  for (int e = 0; e < E; ++e) {
    float di = iv[e] - mi, dl = lv[e] - ml;
    viv += di * di; vlv += dl * dl;
  }
  viv /= E; vlv /= E;
  float bl = 0.01f * (viv / (mi * mi + 1e-10f) + vlv / (ml * ml + 1e-10f));
  tail[0] = bl;
#pragma unroll
  for (int e = 0; e < E; ++e) { tail[1 + e] = lv[e]; tail[9 + e] = iv[e]; }
}

// ---------------- host launcher ----------------
extern "C" void kernel_launch(void* const* d_in, const int* in_sizes, int n_in,
                              void* d_out, int out_size, void* d_ws, size_t ws_size,
                              hipStream_t stream) {
  const float* x  = (const float*)d_in[0];
  const float* wg = (const float*)d_in[1];
  const float* w1 = (const float*)d_in[2];
  const float* w3 = (const float*)d_in[3];
  const float* w2 = (const float*)d_in[4];
  const int N = in_sizes[0] / D;      // 8192 tokens

  float* y    = (float*)d_out;
  float* tail = y + (size_t)N * D;    // [loss, load[8], importance[8]]

  // workspace carve-out
  char* ws = (char*)d_ws;
  size_t off = 0;
  auto carve = [&](size_t bytes) -> void* {
    void* p = ws + off;
    off = (off + bytes + 255) & ~(size_t)255;
    return p;
  };
  int*   cnt  = (int*)carve(16 * 4);          // cnt[8] + imp[8] contiguous
  float* imp  = (float*)((char*)cnt + E * 4);
  int*   hoff = (int*)carve(E * 4);
  int*   atok = (int*)carve((size_t)E * N * 4);
  float* asc  = (float*)carve((size_t)E * N * 4);
  _Float16* xh  = (_Float16*)carve((size_t)N * D * 2);
  _Float16* w1t = (_Float16*)carve((size_t)E * F * D * 2);
  _Float16* w3t = (_Float16*)carve((size_t)E * F * D * 2);
  _Float16* w2t = (_Float16*)carve((size_t)E * D * F * 2);
  _Float16* hb  = (_Float16*)carve(((size_t)2 * N + E * MT) * F * 2); // padded h rows

  // zero accumulator output and gate counters
  fill0_kernel<<<1024, 256, 0, stream>>>((unsigned int*)y, (size_t)N * D);
  fill0_kernel<<<1, 64, 0, stream>>>((unsigned int*)cnt, 16);

  // f32 -> f16 conversions (weights transposed so WMMA B-frags are contiguous)
  cvt_f16_kernel<<<2048, 256, 0, stream>>>(x, xh, (size_t)N * D);
  dim3 tb(32, 8);
  transpose_cvt_kernel<<<dim3(F / 32, D / 32, E), tb, 0, stream>>>(w1, w1t, D, F); // -> [E][F][D]
  transpose_cvt_kernel<<<dim3(F / 32, D / 32, E), tb, 0, stream>>>(w3, w3t, D, F); // -> [E][F][D]
  transpose_cvt_kernel<<<dim3(D / 32, F / 32, E), tb, 0, stream>>>(w2, w2t, F, D); // -> [E][D][F]

  // top-2 gating + expert token lists, then padded prefix offsets
  gate_kernel<<<(N + 3) / 4, 128, 0, stream>>>(x, wg, cnt, imp, atok, asc, N);
  prefix_kernel<<<1, 32, 0, stream>>>(cnt, hoff);

  // routed expert GEMMs
  size_t smem = (size_t)MT * XS * sizeof(_Float16);   // ~257 KB of 320 KB WGP LDS
  (void)hipFuncSetAttribute((const void*)moe_kernel,
                            hipFuncAttributeMaxDynamicSharedMemorySize, (int)smem);
  moe_kernel<<<dim3((N + MT - 1) / MT, E), 256, smem, stream>>>(
      xh, w1t, w3t, w2t, atok, asc, cnt, hoff, hb, y, N);

  // loss + aux tail
  finalize_kernel<<<1, 32, 0, stream>>>(cnt, imp, tail);
}
